// GraphDistillOperator_49624052138317
// MI455X (gfx1250) — matrix operand order repaired and checked by reading, not compiled, not run on hardware
//
#include <hip/hip_runtime.h>
#include <math.h>

// ---------------------------------------------------------------------------
// Types for CDNA5 WMMA (wave32): V_WMMA_F32_16X16X32_BF16
// ---------------------------------------------------------------------------
typedef __attribute__((ext_vector_type(16))) __bf16 v16bf;
typedef __attribute__((ext_vector_type(8)))  __bf16 v8bf;
typedef __attribute__((ext_vector_type(8)))  float  v8f;

#define WMMA_BF16(acc, a, b)                                                   \
  (acc) = __builtin_amdgcn_wmma_f32_16x16x32_bf16(                             \
      false, (a), false, (b), (short)0, (acc), false, false)

// ---------------------------------------------------------------------------
// bf16 helpers (RNE), no reliance on __bf16 arithmetic
// ---------------------------------------------------------------------------
__device__ __forceinline__ unsigned short f2bf_rne(float f) {
  unsigned int u = __float_as_uint(f);
  unsigned int lsb = (u >> 16) & 1u;
  u += 0x7fffu + lsb;
  return (unsigned short)(u >> 16);
}
__device__ __forceinline__ float bf2f(unsigned short h) {
  return __uint_as_float(((unsigned int)h) << 16);
}
__device__ __forceinline__ unsigned int pack_bf2(float a, float b) {
  return (unsigned int)f2bf_rne(a) | ((unsigned int)f2bf_rne(b) << 16);
}
__device__ __forceinline__ v16bf cat8(v8bf a, v8bf b) {
  return __builtin_shufflevector(a, b, 0,1,2,3,4,5,6,7,8,9,10,11,12,13,14,15);
}

// A fragment (bf16 16x32, M x K), row-major A with given row stride.
// Lane l<16 holds M=l, K = {k..k+7, k+16..k+23}; lanes 16-31: +8 on K halves.
__device__ __forceinline__ v16bf load_frag_rowA(const unsigned short* __restrict__ A,
                                                int stride, int m0, int k, int lane) {
  int r = lane & 15;
  int off = (lane >> 4) * 8;
  const unsigned short* p = A + (size_t)(m0 + r) * stride + k + off;
  v8bf a = *(const v8bf*)(p);
  v8bf b = *(const v8bf*)(p + 16);
  return cat8(a, b);
}

// B fragment (bf16 32x16, K x N) from a TRANSPOSED matrix BT[N][K].
// Lane l<16 holds N=l, K = {k..k+7, k+16..k+23}; lanes 16-31: +8 on K halves.
__device__ __forceinline__ v16bf load_frag_colB(const unsigned short* __restrict__ BT,
                                                int K, int n0, int k, int lane) {
  int c = lane & 15;
  int off = (lane >> 4) * 8;
  const unsigned short* p = BT + (size_t)(n0 + c) * K + k + off;
  return cat8(*(const v8bf*)(p), *(const v8bf*)(p + 16));
}

// ---------------------------------------------------------------------------
// Dimensions
// ---------------------------------------------------------------------------
#define NN 8192
#define DD 256

// ---------------------------------------------------------------------------
// K0: row reductions over adj -> s1 = deg/nn, s2 = 1/nn
// one wave per row
// ---------------------------------------------------------------------------
__global__ __launch_bounds__(256) void reduce_kernel(const float* __restrict__ adj,
                                                     float* __restrict__ s1,
                                                     float* __restrict__ s2) {
  int row  = blockIdx.x * 8 + (threadIdx.x >> 5);
  int lane = threadIdx.x & 31;
  const float* p = adj + (size_t)row * NN;
  float sum = 0.0f, mx = 0.0f;
  for (int c = lane; c < NN; c += 32) {
    float v = p[c];
    sum += v;
    mx = fmaxf(mx, v);
  }
  for (int o = 16; o > 0; o >>= 1) {
    sum += __shfl_xor(sum, o, 32);
    mx = fmaxf(mx, __shfl_xor(mx, o, 32));
  }
  if (lane == 0) {
    float nn = sum + (1.0f - mx);
    s1[row] = sum / nn;
    s2[row] = 1.0f / nn;
  }
}

// ---------------------------------------------------------------------------
// K1: split features into hi/lo bf16, row-major AND transposed (for B operand)
// ---------------------------------------------------------------------------
__global__ __launch_bounds__(256) void splitF_kernel(const float* __restrict__ F,
                                                     unsigned short* __restrict__ Fhi,
                                                     unsigned short* __restrict__ Flo,
                                                     unsigned short* __restrict__ FThi,
                                                     unsigned short* __restrict__ FTlo) {
  int row = blockIdx.x;     // node
  int col = threadIdx.x;    // feature dim
  float x = F[(size_t)row * DD + col];
  unsigned short hi = f2bf_rne(x);
  unsigned short lo = f2bf_rne(x - bf2f(hi));
  size_t rm = (size_t)row * DD + col;
  size_t tm = (size_t)col * NN + row;
  Fhi[rm] = hi; Flo[rm] = lo;
  FThi[tm] = hi; FTlo[tm] = lo;
}

// ---------------------------------------------------------------------------
// K1b: transpose+split a 256x256 f32 weight matrix W[k][n] -> WT[n][k] hi/lo
// ---------------------------------------------------------------------------
__global__ __launch_bounds__(256) void transposeW_kernel(const float* __restrict__ W,
                                                         unsigned short* __restrict__ Thi,
                                                         unsigned short* __restrict__ Tlo) {
  int gid = blockIdx.x * 256 + threadIdx.x;
  int n = gid >> 8;
  int k = gid & 255;
  float x = W[(size_t)k * DD + n];
  unsigned short hi = f2bf_rne(x);
  unsigned short lo = f2bf_rne(x - bf2f(hi));
  Thi[(size_t)n * DD + k] = hi;
  Tlo[(size_t)n * DD + k] = lo;
}

// ---------------------------------------------------------------------------
// K2: agg = adj @ F via bf16 WMMA.
//   adj (exact in bf16) staged through LDS once per 16-row strip;
//   F supplied as transposed hi/lo splits (B operand).
//   Output written as row-major hi/lo bf16 splits (only form needed later).
// Block: 128 threads = 4 waves; block computes 16 rows x 256 cols.
// ---------------------------------------------------------------------------
__global__ __launch_bounds__(128) void gemm_adj_kernel(const float* __restrict__ adj,
                                                       const unsigned short* __restrict__ FThi,
                                                       const unsigned short* __restrict__ FTlo,
                                                       unsigned short* __restrict__ aggHi,
                                                       unsigned short* __restrict__ aggLo) {
  __shared__ unsigned short ldsA[16][136];   // 16 rows x 128 K, padded (136*2=272B=17*16)

  int m0   = blockIdx.x * 16;
  int wave = threadIdx.x >> 5;
  int lane = threadIdx.x & 31;
  int nbase = wave * 64;

  int trow = threadIdx.x >> 3;         // 0..15 staging row
  int tcol = (threadIdx.x & 7) * 16;   // 0..112 staging col base

  const v8f vzero = {0.f,0.f,0.f,0.f,0.f,0.f,0.f,0.f};
  v8f acc[4];
#pragma unroll
  for (int t = 0; t < 4; ++t) acc[t] = vzero;

  const float* srcbase = adj + (size_t)(m0 + trow) * NN + tcol;

  for (int kb = 0; kb < NN; kb += 128) {
    // ---- stage adj[16][kb:kb+128] into LDS as bf16 (exact for 0/1) ----
    const float* src = srcbase + kb;
    float4 f0 = *(const float4*)(src + 0);
    float4 f1 = *(const float4*)(src + 4);
    float4 f2 = *(const float4*)(src + 8);
    float4 f3 = *(const float4*)(src + 12);
    if (kb + 128 < NN) __builtin_prefetch(src + 128, 0, 1);
    uint4 q0 = make_uint4(pack_bf2(f0.x, f0.y), pack_bf2(f0.z, f0.w),
                          pack_bf2(f1.x, f1.y), pack_bf2(f1.z, f1.w));
    uint4 q1 = make_uint4(pack_bf2(f2.x, f2.y), pack_bf2(f2.z, f2.w),
                          pack_bf2(f3.x, f3.y), pack_bf2(f3.z, f3.w));
    *(uint4*)&ldsA[trow][tcol]     = q0;
    *(uint4*)&ldsA[trow][tcol + 8] = q1;
    __syncthreads();

    // ---- 4 k-steps of 32 over the staged chunk ----
#pragma unroll
    for (int ks = 0; ks < 128; ks += 32) {
      int r   = lane & 15;
      int off = (lane >> 4) * 8;
      v8bf a0 = *(const v8bf*)&ldsA[r][ks + off];
      v8bf a1 = *(const v8bf*)&ldsA[r][ks + 16 + off];
      v16bf af = cat8(a0, a1);
      int kg = kb + ks;
#pragma unroll
      for (int t = 0; t < 4; ++t) {
        v16bf bh = load_frag_colB(FThi, NN, nbase + t * 16, kg, lane);
        v16bf bl = load_frag_colB(FTlo, NN, nbase + t * 16, kg, lane);
        WMMA_BF16(acc[t], af, bh);
        WMMA_BF16(acc[t], af, bl);
      }
    }
    __syncthreads();
  }

  // ---- epilogue: split-store agg ----
  int nlo = lane & 15;
  int mo  = (lane >> 4) * 8;
#pragma unroll
  for (int t = 0; t < 4; ++t) {
    int n = nbase + t * 16 + nlo;
#pragma unroll
    for (int r = 0; r < 8; ++r) {
      int m = m0 + r + mo;
      float x = acc[t][r];
      unsigned short hi = f2bf_rne(x);
      unsigned short lo = f2bf_rne(x - bf2f(hi));
      aggHi[(size_t)m * DD + n] = hi;
      aggLo[(size_t)m * DD + n] = lo;
    }
  }
}

// ---------------------------------------------------------------------------
// K3: fused mid layer.
//   P = F @ W1,  Q = agg @ W2,  R = agg @ W_agg      (all split-bf16 WMMA)
//   t1 = F - (s1*P + s2*Q + b_distill)   -> hi/lo splits
//   t2 = F + (R + b_agg)                 -> hi/lo splits
// Block: 128 threads = 4 waves; 16 rows x 256 cols per block.
// ---------------------------------------------------------------------------
__global__ __launch_bounds__(128) void mid_kernel(
    const float* __restrict__ F,
    const unsigned short* __restrict__ Fhi, const unsigned short* __restrict__ Flo,
    const unsigned short* __restrict__ Ahi, const unsigned short* __restrict__ Alo,
    const unsigned short* __restrict__ W1Thi, const unsigned short* __restrict__ W1Tlo,
    const unsigned short* __restrict__ W2Thi, const unsigned short* __restrict__ W2Tlo,
    const unsigned short* __restrict__ WaThi, const unsigned short* __restrict__ WaTlo,
    const float* __restrict__ bD, const float* __restrict__ bA,
    const float* __restrict__ s1, const float* __restrict__ s2,
    unsigned short* __restrict__ t1hi, unsigned short* __restrict__ t1lo,
    unsigned short* __restrict__ t2hi, unsigned short* __restrict__ t2lo) {
  int m0   = blockIdx.x * 16;
  int wave = threadIdx.x >> 5;
  int lane = threadIdx.x & 31;
  int nbase = wave * 64;

  const v8f vzero = {0.f,0.f,0.f,0.f,0.f,0.f,0.f,0.f};
  v8f P[4], Q[4], R[4];
#pragma unroll
  for (int t = 0; t < 4; ++t) { P[t] = vzero; Q[t] = vzero; R[t] = vzero; }

#pragma unroll
  for (int k = 0; k < DD; k += 32) {
    v16bf fh = load_frag_rowA(Fhi, DD, m0, k, lane);
    v16bf fl = load_frag_rowA(Flo, DD, m0, k, lane);
    v16bf ah = load_frag_rowA(Ahi, DD, m0, k, lane);
    v16bf al = load_frag_rowA(Alo, DD, m0, k, lane);
#pragma unroll
    for (int t = 0; t < 4; ++t) {
      int n0 = nbase + t * 16;
      v16bf w1h = load_frag_colB(W1Thi, DD, n0, k, lane);
      v16bf w1l = load_frag_colB(W1Tlo, DD, n0, k, lane);
      v16bf w2h = load_frag_colB(W2Thi, DD, n0, k, lane);
      v16bf w2l = load_frag_colB(W2Tlo, DD, n0, k, lane);
      v16bf wah = load_frag_colB(WaThi, DD, n0, k, lane);
      v16bf wal = load_frag_colB(WaTlo, DD, n0, k, lane);
      WMMA_BF16(P[t], fh, w1h); WMMA_BF16(P[t], fh, w1l); WMMA_BF16(P[t], fl, w1h);
      WMMA_BF16(Q[t], ah, w2h); WMMA_BF16(Q[t], ah, w2l); WMMA_BF16(Q[t], al, w2h);
      WMMA_BF16(R[t], ah, wah); WMMA_BF16(R[t], ah, wal); WMMA_BF16(R[t], al, wah);
    }
  }

  int nlo = lane & 15;
  int mo  = (lane >> 4) * 8;
#pragma unroll
  for (int t = 0; t < 4; ++t) {
    int n = nbase + t * 16 + nlo;
    float bd = bD[n];
    float ba = bA[n];
#pragma unroll
    for (int r = 0; r < 8; ++r) {
      int m = m0 + r + mo;
      size_t idx = (size_t)m * DD + n;
      float f  = F[idx];
      float nf = s1[m] * P[t][r] + s2[m] * Q[t][r] + bd;
      float v1 = f - nf;
      float v2 = f + R[t][r] + ba;
      unsigned short h1 = f2bf_rne(v1);
      unsigned short h2 = f2bf_rne(v2);
      t1hi[idx] = h1; t1lo[idx] = f2bf_rne(v1 - bf2f(h1));
      t2hi[idx] = h2; t2lo[idx] = f2bf_rne(v2 - bf2f(h2));
    }
  }
}

// ---------------------------------------------------------------------------
// K4: out = tanh(A @ W + b), split-bf16 WMMA, f32 output.
// ---------------------------------------------------------------------------
__global__ __launch_bounds__(128) void out_gemm_tanh(
    const unsigned short* __restrict__ Ahi, const unsigned short* __restrict__ Alo,
    const unsigned short* __restrict__ BThi, const unsigned short* __restrict__ BTlo,
    const float* __restrict__ bias, float* __restrict__ out) {
  int m0   = blockIdx.x * 16;
  int wave = threadIdx.x >> 5;
  int lane = threadIdx.x & 31;
  int nbase = wave * 64;

  const v8f vzero = {0.f,0.f,0.f,0.f,0.f,0.f,0.f,0.f};
  v8f acc[4];
#pragma unroll
  for (int t = 0; t < 4; ++t) acc[t] = vzero;

#pragma unroll
  for (int k = 0; k < DD; k += 32) {
    v16bf ah = load_frag_rowA(Ahi, DD, m0, k, lane);
    v16bf al = load_frag_rowA(Alo, DD, m0, k, lane);
#pragma unroll
    for (int t = 0; t < 4; ++t) {
      int n0 = nbase + t * 16;
      v16bf bh = load_frag_colB(BThi, DD, n0, k, lane);
      v16bf bl = load_frag_colB(BTlo, DD, n0, k, lane);
      WMMA_BF16(acc[t], ah, bh);
      WMMA_BF16(acc[t], ah, bl);
      WMMA_BF16(acc[t], al, bh);
    }
  }

  int nlo = lane & 15;
  int mo  = (lane >> 4) * 8;
#pragma unroll
  for (int t = 0; t < 4; ++t) {
    int n = nbase + t * 16 + nlo;
    float b = bias[n];
#pragma unroll
    for (int r = 0; r < 8; ++r) {
      int m = m0 + r + mo;
      out[(size_t)m * DD + n] = tanhf(acc[t][r] + b);
    }
  }
}

// ---------------------------------------------------------------------------
// Host-side launcher
// ---------------------------------------------------------------------------
extern "C" void kernel_launch(void* const* d_in, const int* in_sizes, int n_in,
                              void* d_out, int out_size, void* d_ws, size_t ws_size,
                              hipStream_t stream) {
  (void)in_sizes; (void)n_in; (void)out_size; (void)ws_size;

  const float* F   = (const float*)d_in[0];
  const float* adj = (const float*)d_in[1];
  const float* Wd  = (const float*)d_in[2];   // [512,256]
  const float* bd  = (const float*)d_in[3];
  const float* Wdo = (const float*)d_in[4];   // [256,256]
  const float* bdo = (const float*)d_in[5];
  const float* Wag = (const float*)d_in[6];   // [256,256]
  const float* bag = (const float*)d_in[7];
  const float* Wao = (const float*)d_in[8];   // [256,256]
  const float* bao = (const float*)d_in[9];

  float* out0 = (float*)d_out;                       // feature_updated
  float* out1 = (float*)d_out + (size_t)NN * DD;     // feature_aggregate

  // workspace carving
  size_t off = 0;
  char* base = (char*)d_ws;
  auto carve = [&](size_t bytes) -> void* {
    void* p = base + off;
    off += (bytes + 255) & ~(size_t)255;
    return p;
  };
  const size_t ND2 = (size_t)NN * DD * sizeof(unsigned short);  // 4 MB
  const size_t WD2 = (size_t)DD * DD * sizeof(unsigned short);  // 128 KB

  float* s1 = (float*)carve(NN * sizeof(float));
  float* s2 = (float*)carve(NN * sizeof(float));
  unsigned short* Fhi  = (unsigned short*)carve(ND2);
  unsigned short* Flo  = (unsigned short*)carve(ND2);
  unsigned short* FThi = (unsigned short*)carve(ND2);
  unsigned short* FTlo = (unsigned short*)carve(ND2);
  unsigned short* Ahi  = (unsigned short*)carve(ND2);   // agg splits
  unsigned short* Alo  = (unsigned short*)carve(ND2);
  unsigned short* t1hi = (unsigned short*)carve(ND2);
  unsigned short* t1lo = (unsigned short*)carve(ND2);
  unsigned short* t2hi = (unsigned short*)carve(ND2);
  unsigned short* t2lo = (unsigned short*)carve(ND2);
  unsigned short* W1Thi = (unsigned short*)carve(WD2);
  unsigned short* W1Tlo = (unsigned short*)carve(WD2);
  unsigned short* W2Thi = (unsigned short*)carve(WD2);
  unsigned short* W2Tlo = (unsigned short*)carve(WD2);
  unsigned short* WaThi = (unsigned short*)carve(WD2);
  unsigned short* WaTlo = (unsigned short*)carve(WD2);
  unsigned short* WdoThi = (unsigned short*)carve(WD2);
  unsigned short* WdoTlo = (unsigned short*)carve(WD2);
  unsigned short* WaoThi = (unsigned short*)carve(WD2);
  unsigned short* WaoTlo = (unsigned short*)carve(WD2);

  // 1) row reductions over adj
  reduce_kernel<<<NN / 8, 256, 0, stream>>>(adj, s1, s2);

  // 2) split features (row-major + transposed)
  splitF_kernel<<<NN, 256, 0, stream>>>(F, Fhi, Flo, FThi, FTlo);

  // 3) transpose+split weights
  transposeW_kernel<<<DD, 256, 0, stream>>>(Wd, W1Thi, W1Tlo);                 // rows 0..255
  transposeW_kernel<<<DD, 256, 0, stream>>>(Wd + (size_t)DD * DD, W2Thi, W2Tlo); // rows 256..511
  transposeW_kernel<<<DD, 256, 0, stream>>>(Wag, WaThi, WaTlo);
  transposeW_kernel<<<DD, 256, 0, stream>>>(Wdo, WdoThi, WdoTlo);
  transposeW_kernel<<<DD, 256, 0, stream>>>(Wao, WaoThi, WaoTlo);

  // 4) agg = adj @ F (the big one)
  gemm_adj_kernel<<<NN / 16, 128, 0, stream>>>(adj, FThi, FTlo, Ahi, Alo);

  // 5) fused mid layer -> t1, t2 splits
  mid_kernel<<<NN / 16, 128, 0, stream>>>(F, Fhi, Flo, Ahi, Alo,
                                          W1Thi, W1Tlo, W2Thi, W2Tlo, WaThi, WaTlo,
                                          bd, bag, s1, s2,
                                          t1hi, t1lo, t2hi, t2lo);

  // 6) outputs
  out_gemm_tanh<<<NN / 16, 128, 0, stream>>>(t1hi, t1lo, WdoThi, WdoTlo, bdo, out0);
  out_gemm_tanh<<<NN / 16, 128, 0, stream>>>(t2hi, t2lo, WaoThi, WaoTlo, bao, out1);
}
